// BiNCPEncoder_81346680586932
// MI455X (gfx1250) — matrix operand order repaired and checked by reading, not compiled, not run on hardware
//
#include <hip/hip_runtime.h>
#include <hip/hip_fp16.h>

// ---------------------------------------------------------------------------
// BiNCP CfC encoder for gfx1250 (MI455X).
//  kernel 1: pack_kernel  -- fold masks / fuse ta+tb, emit f16 WMMA B-fragments
//  kernel 2: ncp_kernel   -- persistent recurrence: 16 fwd WGs (512 steps),
//                            16 bwd WGs (1 step), WMMA f32_16x16x32_f16
//                            weights LDS-resident (layer0 time-gate streams
//                            from L2 with prefetch; everything else in LDS)
//  kernel 3: fc_kernel    -- final [256x128]@[128x256] + bias
// ---------------------------------------------------------------------------

typedef __attribute__((ext_vector_type(16))) _Float16 v16h;
typedef __attribute__((ext_vector_type(8)))  float    v8f;

union U16h { uint4 u[2]; v16h h; };

// problem constants
constexpr int SEQN = 512;
constexpr int XDIM = 128;

// fragment geometry: one B fragment = 32x16 f16 = 512 halves = 1024B
constexpr int FRAG_H = 512;
constexpr int F0 = 64, F1 = 30, F2 = 20;          // frags per matrix per layer
// packed-weight offsets (halves) inside one direction: [l0 ff1|ff2|t][l1 ...][l2 ...]
constexpr int G0_T  = 2 * F0 * FRAG_H;            // 65536
constexpr int G1_F1 = 3 * F0 * FRAG_H;            // 98304
constexpr int G2_F1 = G1_F1 + 3 * F1 * FRAG_H;    // 144384
constexpr int DIR_H = G2_F1 + 3 * F2 * FRAG_H;    // 175104 halves per direction

// LDS layout (halves): l0 ff1/ff2, l1 ff1/ff2/t, l2 ff1/ff2/t, acts, biases
constexpr int L0_F1 = 0;
constexpr int L0_F2 = F0 * FRAG_H;                // 32768
constexpr int L1_F1 = 2 * F0 * FRAG_H;            // 65536
constexpr int L1_F2 = L1_F1 + F1 * FRAG_H;        // 80896
constexpr int L1_T  = L1_F1 + 2 * F1 * FRAG_H;    // 96256
constexpr int L2_F1 = L1_F1 + 3 * F1 * FRAG_H;    // 111616
constexpr int L2_F2 = L2_F1 + F2 * FRAG_H;        // 121856
constexpr int L2_T  = L2_F1 + 2 * F2 * FRAG_H;    // 132096
constexpr int WLDS_H = L2_F1 + 3 * F2 * FRAG_H;   // 142336 halves = 284672B
constexpr int ACT0_H = WLDS_H;                    // 16 x 256 halves
constexpr int ACT1_H = ACT0_H + 16 * 256;         // 16 x 192
constexpr int ACT2_H = ACT1_H + 16 * 192;         // 16 x 160
constexpr int ACT_END_H = ACT2_H + 16 * 160;      // 152064 halves
constexpr int BIAS_OFF_B = ACT_END_H * 2;         // 304128 bytes
constexpr size_t SMEM_BYTES = BIAS_OFF_B + 816 * 4; // 307392 bytes (< 320KB)

// workspace byte offsets
constexpr size_t WS_WPAK  = 0;                         // 2*175104*2 = 700416 B
constexpr size_t WS_BPAK  = 700416;                    // 2*816*4    = 6528 B
constexpr size_t WS_MOTF  = 706944;                    // 256*64*4   = 65536 B
constexpr size_t WS_MOTB  = 772480;                    // 256*64*4

struct PackArgs {
  const float* W[2][3][4];   // [dir][layer][ff1,ff2,ta,tb]
  const float* B[2][3][4];   // biases, same order
  const float* M[2][3];      // masks
};

__device__ __forceinline__ float fast_rcp(float x)     { return __builtin_amdgcn_rcpf(x); }
__device__ __forceinline__ float fast_sigmoid(float x) { return fast_rcp(1.0f + __expf(-x)); }
__device__ __forceinline__ float fast_tanh(float x)    { return 1.0f - 2.0f * fast_rcp(__expf(2.0f * x) + 1.0f); }

// ---------------------------------------------------------------------------
// pack_kernel: build WMMA B-fragments (f16) + fused biases (f32)
// Fragment layout per ISA 7.12.2 (16-bit B 32x16): lanes 0-15 hold column
// N=lane, K={0..7,16..23}; lanes 16-31 hold N=lane-16, K={8..15,24..31}.
// ---------------------------------------------------------------------------
__global__ void pack_kernel(PackArgs pa, _Float16* __restrict__ wout,
                            float* __restrict__ bout) {
  constexpr int FRAG_TOT = 2 * DIR_H;   // 350208 halves (both dirs)
  constexpr int BIAS_TOT = 2 * 816;
  const int gid = blockIdx.x * blockDim.x + threadIdx.x;
  if (gid < FRAG_TOT) {
    const int d = gid / DIR_H;
    const int r = gid % DIR_H;
    const int frag = r / FRAG_H;
    const int rem  = r % FRAG_H;
    const int lane = rem >> 4, half = rem & 15;
    int l, m, f;
    if (frag < 192)      { l = 0; m = frag >> 6;         f = frag & 63; }
    else if (frag < 282) { l = 1; m = (frag - 192) / F1; f = (frag - 192) % F1; }
    else                 { l = 2; m = (frag - 282) / F2; f = (frag - 282) % F2; }
    const int KT_[3] = {8, 6, 5};
    const int FI[3]  = {244, 192, 140};
    const int UN[3]  = {116, 76, 64};
    const int kt = f % KT_[l], nt = f / KT_[l];
    const int k = kt * 32 + ((lane >= 16) ? 8 : 0) + (half & 7) + ((half >= 8) ? 16 : 0);
    const int n = nt * 16 + (lane & 15);
    float v = 0.0f;
    if (k < FI[l] && n < UN[l]) {
      const int idx = k * UN[l] + n;
      if (m < 2) v = pa.W[d][l][m][idx] * pa.M[d][l][idx];
      else       v = pa.W[d][l][2][idx] + pa.W[d][l][3][idx];
    }
    wout[gid] = (_Float16)v;
  } else if (gid < FRAG_TOT + BIAS_TOT) {
    const int e0 = gid - FRAG_TOT;
    const int d = e0 / 816;
    const int e = e0 % 816;
    const int UN[3] = {116, 76, 64};
    int l, m, n;
    if (e < 384)      { l = 0; m = e / 128;         n = e % 128; }
    else if (e < 624) { l = 1; m = (e - 384) / 80;  n = (e - 384) % 80; }
    else              { l = 2; m = (e - 624) / 64;  n = (e - 624) % 64; }
    float v = 0.0f;
    if (n < UN[l]) {
      if (m == 0)      v = pa.B[d][l][0][n];
      else if (m == 1) v = pa.B[d][l][1][n];
      else             v = pa.B[d][l][2][n] + pa.B[d][l][3][n];
    }
    bout[e0] = v;
  }
}

// ---------------------------------------------------------------------------
// One CfC layer step for a 16-row batch tile.
//  A fragments (16x32 f16) read from LDS act buffer per ISA A layout:
//   lane<16: M=lane, K chunks {kt*32+0..7, kt*32+16..23}
//   lane>=16: M=lane-16, K chunks {+8..15, +24..31}
//  ff1/ff2 B fragments from LDS; time-gate fragments from LDS (layers 1,2)
//  or streamed from L2 with prefetch (layer 0, WT_GLB=true).
// ---------------------------------------------------------------------------
template <int KT, int NT, int KPAD, int UNITS_L, bool WT_GLB>
__device__ __forceinline__ void layer_step(
    int tid, const _Float16* __restrict__ actIn,
    const _Float16* __restrict__ wf1, const _Float16* __restrict__ wf2,
    const _Float16* __restrict__ wtg, const float* __restrict__ bias,
    _Float16* __restrict__ dstA, int strideA,
    _Float16* __restrict__ dstB, int strideB, float* __restrict__ motor) {
  const int wv = tid >> 5, lane = tid & 31;
  const int nlo = lane & 15;
  const int hi  = lane >> 4;
  _Float16 hv[8];
  if (wv < NT) {                       // wave-uniform: EXEC all-ones inside
    v8f a1 = {}, a2 = {}, at = {};
    #pragma unroll
    for (int kt = 0; kt < KT; ++kt) {
      // A fragment from activations in LDS
      const _Float16* ap = actIn + nlo * KPAD + kt * 32 + hi * 8;
      U16h A;
      A.u[0] = *(const uint4*)ap;
      A.u[1] = *(const uint4*)(ap + 16);
      const size_t fo = (size_t)(wv * KT + kt) * FRAG_H + lane * 16;
      U16h B1, B2, BT;
      { const uint4* p = (const uint4*)(wf1 + fo); B1.u[0] = p[0]; B1.u[1] = p[1]; }
      { const uint4* p = (const uint4*)(wf2 + fo); B2.u[0] = p[0]; B2.u[1] = p[1]; }
      { const uint4* p = (const uint4*)(wtg + fo); BT.u[0] = p[0]; BT.u[1] = p[1]; }
      if (WT_GLB && (kt + 1 < KT)) __builtin_prefetch(wtg + fo + FRAG_H, 0, 1);
      a1 = __builtin_amdgcn_wmma_f32_16x16x32_f16(false, A.h, false, B1.h, (short)0, a1, false, false);
      a2 = __builtin_amdgcn_wmma_f32_16x16x32_f16(false, A.h, false, B2.h, (short)0, a2, false, false);
      at = __builtin_amdgcn_wmma_f32_16x16x32_f16(false, A.h, false, BT.h, (short)0, at, false, false);
    }
    const int n = wv * 16 + nlo;
    const bool valid = (n < UNITS_L);
    const float b1 = bias[n], b2 = bias[NT * 16 + n], bt = bias[2 * NT * 16 + n];
    #pragma unroll
    for (int r = 0; r < 8; ++r) {
      const float s  = fast_sigmoid(at[r] + bt);
      const float f1 = fast_tanh(a1[r] + b1);
      const float f2 = fast_tanh(a2[r] + b2);
      const float hv_f = valid ? (f1 + s * (f2 - f1)) : 0.0f;
      hv[r] = (_Float16)hv_f;
    }
  }
  __syncthreads();                     // all readers of actIn done
  if (wv < NT) {
    const int n = wv * 16 + nlo;
    if (n < UNITS_L) {
      const int mb = hi * 8;           // D layout: lane<16 -> M 0..7, else 8..15
      #pragma unroll
      for (int r = 0; r < 8; ++r) {
        dstA[(mb + r) * strideA + n] = hv[r];
        if (dstB)  dstB[(mb + r) * strideB + n] = hv[r];
        if (motor) motor[(mb + r) * 64 + n] = (float)hv[r];
      }
    }
  }
  __syncthreads();                     // writes visible before next layer
}

__device__ __forceinline__ void cp_u4(int tid, uint4* dst, const uint4* src, int n) {
  for (int i = tid; i < n; i += 256) dst[i] = src[i];
}

// ---------------------------------------------------------------------------
// Persistent recurrence kernel: blocks 0..15 forward (512 steps),
// blocks 16..31 backward (1 step on x[:, 511] from zero state).
// ---------------------------------------------------------------------------
__global__ __launch_bounds__(256, 1) void ncp_kernel(
    const float* __restrict__ x, const _Float16* __restrict__ wpak,
    const float* __restrict__ bpak, float* __restrict__ motor_f,
    float* __restrict__ motor_b) {
  extern __shared__ __align__(16) char smem[];
  _Float16* wlds  = (_Float16*)smem;
  _Float16* act0  = wlds + ACT0_H;
  _Float16* act1  = wlds + ACT1_H;
  _Float16* act2  = wlds + ACT2_H;
  float*    biasL = (float*)(smem + BIAS_OFF_B);

  const int tid = threadIdx.x;
  const int blk = blockIdx.x;
  const bool fwd = (blk < 16);
  const int bt = fwd ? blk : blk - 16;
  const _Float16* wdir = wpak + (fwd ? 0 : DIR_H);
  const float*    bdir = bpak + (fwd ? 0 : 816);
  float* motor = (fwd ? motor_f : motor_b) + (size_t)bt * 16 * 64;
  const int nsteps = fwd ? SEQN : 1;

  // stage weights into LDS: l0 ff1/ff2; l1 ff1/ff2/t; l2 ff1/ff2/t
  cp_u4(tid, (uint4*)(wlds + L0_F1), (const uint4*)(wdir),          2 * F0 * FRAG_H / 8);
  cp_u4(tid, (uint4*)(wlds + L1_F1), (const uint4*)(wdir + G1_F1),  3 * F1 * FRAG_H / 8);
  cp_u4(tid, (uint4*)(wlds + L2_F1), (const uint4*)(wdir + G2_F1),  3 * F2 * FRAG_H / 8);
  // biases
  for (int i = tid; i < 816; i += 256) biasL[i] = bdir[i];
  // zero activation buffers (h = 0, padding = 0 forever)
  {
    const uint4 z = {0u, 0u, 0u, 0u};
    uint4* a = (uint4*)act0;
    for (int i = tid; i < (ACT_END_H - ACT0_H) / 8; i += 256) a[i] = z;
  }
  __syncthreads();

  for (int step = 0; step < nsteps; ++step) {
    const int t = fwd ? step : (SEQN - 1);
    // stage x_t tile: 16 rows x 128 f32 -> f16 into act0 cols [0,128)
    {
      const int m = tid >> 4;
      const int c = (tid & 15) * 8;
      const float* xp = x + (((size_t)(bt * 16 + m)) * SEQN + t) * XDIM + c;
      const float4 v0 = *(const float4*)xp;
      const float4 v1 = *(const float4*)(xp + 4);
      _Float16* dp = act0 + m * 256 + c;
      dp[0] = (_Float16)v0.x; dp[1] = (_Float16)v0.y;
      dp[2] = (_Float16)v0.z; dp[3] = (_Float16)v0.w;
      dp[4] = (_Float16)v1.x; dp[5] = (_Float16)v1.y;
      dp[6] = (_Float16)v1.z; dp[7] = (_Float16)v1.w;
    }
    __syncthreads();
    const bool last = (step == nsteps - 1);
    // layer 0 (inter):  xh=[x(128)|h0(116)|pad] K=256, N=128 (116 valid)
    layer_step<8, 8, 256, 116, true>(tid, act0, wlds + L0_F1, wlds + L0_F2,
                                     wdir + G0_T, biasL,
                                     act0 + 128, 256, act1, 192, nullptr);
    // layer 1 (command): xh=[h0(116)|h1(76)] K=192, N=80 (76 valid)
    layer_step<6, 5, 192, 76, false>(tid, act1, wlds + L1_F1, wlds + L1_F2,
                                     wlds + L1_T, biasL + 384,
                                     act1 + 116, 192, act2, 160, nullptr);
    // layer 2 (motor):  xh=[h1(76)|h2(64)|pad] K=160, N=64
    layer_step<5, 4, 160, 64, false>(tid, act2, wlds + L2_F1, wlds + L2_F2,
                                     wlds + L2_T, biasL + 624,
                                     act2 + 76, 160, nullptr, 0,
                                     last ? motor : nullptr);
  }
}

// ---------------------------------------------------------------------------
// fc: hidden[b][j] = sum_i cat[b][i] * fc_w[i][j] + fc_b[j]
// ---------------------------------------------------------------------------
__global__ __launch_bounds__(256) void fc_kernel(
    const float* __restrict__ motor_f, const float* __restrict__ motor_b,
    const float* __restrict__ fc_w, const float* __restrict__ fc_b,
    float* __restrict__ out) {
  __shared__ float cat[128];
  const int b = blockIdx.x, j = threadIdx.x;
  if (j < 64)        cat[j] = motor_f[b * 64 + j];
  else if (j < 128)  cat[j] = motor_b[b * 64 + (j - 64)];
  __syncthreads();
  float acc = fc_b[j];
  #pragma unroll 8
  for (int i = 0; i < 128; ++i) acc = fmaf(cat[i], fc_w[i * 256 + j], acc);
  out[b * 256 + j] = acc;
}

// ---------------------------------------------------------------------------
extern "C" void kernel_launch(void* const* d_in, const int* in_sizes, int n_in,
                              void* d_out, int out_size, void* d_ws, size_t ws_size,
                              hipStream_t stream) {
  (void)in_sizes; (void)n_in; (void)out_size; (void)ws_size;
  // input order: x, params_f(3x[Wff1,bff1,Wff2,bff2,Wta,bta,Wtb,btb]),
  //              params_b(same), masks_f(3), masks_b(3), fc_w, fc_b
  const float* x = (const float*)d_in[0];
  PackArgs pa;
  for (int d = 0; d < 2; ++d) {
    const int base = 1 + d * 24;
    for (int l = 0; l < 3; ++l) {
      const int lb = base + l * 8;
      pa.W[d][l][0] = (const float*)d_in[lb + 0];  // W_ff1
      pa.B[d][l][0] = (const float*)d_in[lb + 1];
      pa.W[d][l][1] = (const float*)d_in[lb + 2];  // W_ff2
      pa.B[d][l][1] = (const float*)d_in[lb + 3];
      pa.W[d][l][2] = (const float*)d_in[lb + 4];  // W_ta
      pa.B[d][l][2] = (const float*)d_in[lb + 5];
      pa.W[d][l][3] = (const float*)d_in[lb + 6];  // W_tb
      pa.B[d][l][3] = (const float*)d_in[lb + 7];
      pa.M[d][l]    = (const float*)d_in[49 + d * 3 + l];
    }
  }
  const float* fc_w = (const float*)d_in[55];
  const float* fc_b = (const float*)d_in[56];

  char* ws = (char*)d_ws;
  _Float16* wpak   = (_Float16*)(ws + WS_WPAK);
  float*    bpak   = (float*)(ws + WS_BPAK);
  float*    motorF = (float*)(ws + WS_MOTF);
  float*    motorB = (float*)(ws + WS_MOTB);

  const int NTOT = 2 * DIR_H + 2 * 816;
  pack_kernel<<<(NTOT + 255) / 256, 256, 0, stream>>>(pa, wpak, bpak);
  ncp_kernel<<<32, 256, SMEM_BYTES, stream>>>(x, wpak, bpak, motorF, motorB);
  fc_kernel<<<256, 256, 0, stream>>>(motorF, motorB, fc_w, fc_b, (float*)d_out);
}
